// FullAttention_25898652795051
// MI455X (gfx1250) — matrix operand order repaired and checked by reading, not compiled, not run on hardware
//
#include <hip/hip_runtime.h>
#include <math.h>

// ---------------------------------------------------------------------------
// Full attention fwd (returns V and A) for B=4, L=S=2048, H=8, E=64, fp32.
// Strategy: f16 WMMA (f32 accum) for both GEMMs, f32 softmax, scores staged
// in LDS per 16-row L tile. One workgroup = (b, h, 16 rows), 4 waves.
// ---------------------------------------------------------------------------

typedef __attribute__((ext_vector_type(16))) _Float16 v16h;
typedef __attribute__((ext_vector_type(8)))  _Float16 v8h;
typedef __attribute__((ext_vector_type(8)))  float    v8f;

#define Bdim   4
#define Ldim   2048
#define Sdim   2048
#define Hdim   8
#define Edim   64
#define LTILE  16
#define NWAVES 4
#define NTHREADS (NWAVES * 32)
#define PSTRIDE 2064            // 2048 + 16 half pad -> 4128B row stride (bank spread)
#define SCALE  0.125f           // 1/sqrt(64)

__global__ __launch_bounds__(NTHREADS)
void attn_fwd_kernel(const float* __restrict__ q,
                     const float* __restrict__ k,
                     const float* __restrict__ v,
                     const unsigned char* __restrict__ mask,  // bool bytes [B,L,S]
                     float* __restrict__ outV,                // [B,L,H,E]
                     float* __restrict__ outA)                // [B,H,L,S]
{
    __shared__ alignas(16) _Float16 P[LTILE * PSTRIDE];   // scores -> exp probs (f16)
    __shared__ float red[LTILE][8];                       // per-row partial reductions
    __shared__ float inv_sum[LTILE];

    const int tid  = threadIdx.x;
    const int wave = tid >> 5;
    const int lane = tid & 31;
    const int li   = lane & 15;   // low-lane index (row / col within 16)
    const int hi   = lane >> 4;   // half-wave select

    const int nlt = Ldim / LTILE;                 // 128 L tiles per (b,h)
    const int lt  = blockIdx.x % nlt;             // innermost -> L2 reuse of K/V
    const int bh  = blockIdx.x / nlt;
    const int h   = bh % Hdim;
    const int b   = bh / Hdim;
    const int l0  = lt * LTILE;

    // ---------------- Phase 1: raw scores = Q * K^T  (f16 WMMA, f32 acc) ----
    // A fragments (Q rows), contiguous k order (matches B):
    //   halves[0..7]  = Q[row][kk*32 + hi*16 + j]
    //   halves[8..15] = Q[row][kk*32 + hi*16 + 8 + j]
    v16h aq[2];
    {
        const float* qrow = q + (((size_t)b * Ldim + (l0 + li)) * Hdim + h) * Edim;
        #pragma unroll
        for (int kk = 0; kk < 2; ++kk) {
            const float4* p0 = (const float4*)(qrow + kk * 32 + hi * 16);
            const float4* p1 = (const float4*)(qrow + kk * 32 + hi * 16 + 8);
            #pragma unroll
            for (int jj = 0; jj < 2; ++jj) {
                float4 f0 = p0[jj], f1 = p1[jj];
                aq[kk][jj*4+0] = (_Float16)f0.x; aq[kk][jj*4+1] = (_Float16)f0.y;
                aq[kk][jj*4+2] = (_Float16)f0.z; aq[kk][jj*4+3] = (_Float16)f0.w;
                aq[kk][8+jj*4+0] = (_Float16)f1.x; aq[kk][8+jj*4+1] = (_Float16)f1.y;
                aq[kk][8+jj*4+2] = (_Float16)f1.z; aq[kk][8+jj*4+3] = (_Float16)f1.w;
            }
        }
    }

    for (int st = wave; st < Sdim / 16; st += NWAVES) {
        const int s0 = st * 16;
        v8f c = {};
        #pragma unroll
        for (int kk = 0; kk < 2; ++kk) {
            // B fragment: column N = li is K-row s0+li; lane holds e = kk*32 + hi*16 + [0..15]
            v16h bf;
            const float4* kp = (const float4*)(k +
                (((size_t)b * Sdim + (s0 + li)) * Hdim + h) * Edim + kk * 32 + hi * 16);
            #pragma unroll
            for (int jj = 0; jj < 4; ++jj) {
                float4 f = kp[jj];
                bf[jj*4+0] = (_Float16)f.x; bf[jj*4+1] = (_Float16)f.y;
                bf[jj*4+2] = (_Float16)f.z; bf[jj*4+3] = (_Float16)f.w;
            }
            c = __builtin_amdgcn_wmma_f32_16x16x32_f16(false, aq[kk], false, bf,
                                                       (short)0, c, false, false);
        }
        // C layout: element r -> row (r + hi*8), col (s0 + li). Store raw score f16.
        #pragma unroll
        for (int r = 0; r < 8; ++r) {
            const int row = r + hi * 8;
            P[row * PSTRIDE + s0 + li] = (_Float16)c[r];
        }
    }
    __syncthreads();

    // ---------------- Phase 2: softmax (f32), write A, keep exp in LDS ------
    const int srow = tid >> 3;          // 0..15
    const int seg  = tid & 7;           // 8 threads per row
    const int SEG  = Sdim / 8;          // 256 cols per thread
    const int c0   = seg * SEG;
    const unsigned char* mrow = mask + ((size_t)b * Ldim + (l0 + srow)) * Sdim + c0;
    _Float16* prow = P + srow * PSTRIDE + c0;

    float m = -3.0e38f;
    for (int c = 0; c < SEG; ++c) {
        float s = mrow[c] ? -10000.0f : (float)prow[c] * SCALE;
        m = fmaxf(m, s);
    }
    red[srow][seg] = m;
    __syncthreads();
    float rm = red[srow][0];
    #pragma unroll
    for (int i = 1; i < 8; ++i) rm = fmaxf(rm, red[srow][i]);

    float sum = 0.0f;
    for (int c = 0; c < SEG; ++c) {
        float s = mrow[c] ? -10000.0f : (float)prow[c] * SCALE;
        float e = __expf(s - rm);
        sum += e;
        prow[c] = (_Float16)e;          // unnormalized prob, f16
    }
    __syncthreads();                    // all reads of red (max) done
    red[srow][seg] = sum;
    __syncthreads();
    float rs = red[srow][0];
    #pragma unroll
    for (int i = 1; i < 8; ++i) rs += red[srow][i];
    const float inv = 1.0f / rs;
    if (seg == 0) inv_sum[srow] = inv;

    // write normalized A (f32, float4 stores, contiguous per thread)
    float* arow = outA + (((size_t)(b * Hdim + h)) * Ldim + (l0 + srow)) * Sdim + c0;
    for (int c4 = 0; c4 < SEG / 4; ++c4) {
        float4 o;
        o.x = (float)prow[c4*4+0] * inv;
        o.y = (float)prow[c4*4+1] * inv;
        o.z = (float)prow[c4*4+2] * inv;
        o.w = (float)prow[c4*4+3] * inv;
        ((float4*)arow)[c4] = o;
    }
    __syncthreads();                    // P + inv_sum visible to all waves

    // ---------------- Phase 3: out = P * V  (f16 WMMA, f32 acc) -------------
    // Each wave owns one 16x16 output tile: columns e0..e0+15.
    const int e0 = wave * 16;
    v8f acc = {};
    for (int kt = 0; kt < Sdim / 32; ++kt) {
        const int k0 = kt * 32;
        // A fragment from LDS probs: row li, K = hi*16 + [0..15] (contiguous, matches B)
        v16h af;
        {
            const v8h* p0 = (const v8h*)(P + li * PSTRIDE + k0 + hi * 16);
            const v8h* p1 = (const v8h*)(P + li * PSTRIDE + k0 + hi * 16 + 8);
            v8h x0 = *p0, x1 = *p1;
            #pragma unroll
            for (int j = 0; j < 8; ++j) { af[j] = x0[j]; af[j + 8] = x1[j]; }
        }
        // B fragment from V: column N = li -> e = e0+li; K = hi*16 + j -> s = k0 + ...
        v16h bf;
        const float* vcol = v + (((size_t)b * Sdim + (k0 + hi * 16)) * Hdim + h) * Edim
                              + e0 + li;
        #pragma unroll
        for (int j = 0; j < 16; ++j)
            bf[j] = (_Float16)vcol[(size_t)j * Hdim * Edim];
        acc = __builtin_amdgcn_wmma_f32_16x16x32_f16(false, af, false, bf,
                                                     (short)0, acc, false, false);
    }
    // scale by 1/rowsum and store
    #pragma unroll
    for (int r = 0; r < 8; ++r) {
        const int row = r + hi * 8;
        outV[(((size_t)b * Ldim + (l0 + row)) * Hdim + h) * Edim + e0 + li] =
            acc[r] * inv_sum[row];
    }
}

extern "C" void kernel_launch(void* const* d_in, const int* in_sizes, int n_in,
                              void* d_out, int out_size, void* d_ws, size_t ws_size,
                              hipStream_t stream)
{
    (void)in_sizes; (void)n_in; (void)out_size; (void)d_ws; (void)ws_size;
    const float*         q    = (const float*)d_in[0];
    const float*         k    = (const float*)d_in[1];
    const float*         v    = (const float*)d_in[2];
    const unsigned char* mask = (const unsigned char*)d_in[3];   // bool bytes

    float* outV = (float*)d_out;                                   // [B,L,H,E]
    float* outA = outV + (size_t)Bdim * Ldim * Hdim * Edim;        // [B,H,L,S]

    dim3 grid(Bdim * Hdim * (Ldim / LTILE));   // 4096 workgroups
    attn_fwd_kernel<<<grid, NTHREADS, 0, stream>>>(q, k, v, mask, outV, outA);
}
